// EncoderDecoder_66159676227911
// MI455X (gfx1250) — matrix operand order repaired and checked by reading, not compiled, not run on hardware
//
#include <hip/hip_runtime.h>
#include <hip/hip_bf16.h>

// ---------------------------------------------------------------------------
// Problem constants (match reference)
// ---------------------------------------------------------------------------
#define H      1024
#define EDIM   100
#define VOUT   50257
#define VPAD   50304          // VOUT rounded up to 128 (16 rows/wave * 8 waves/block)
#define SEQL   64

typedef __attribute__((ext_vector_type(16))) __bf16          v16bf;
typedef __attribute__((ext_vector_type(8)))  float           v8f;
typedef __attribute__((ext_vector_type(4)))  float           f32x4;
typedef __attribute__((ext_vector_type(4)))  unsigned short  u16x4;

// float -> bf16 bits, round-to-nearest-even
__device__ __forceinline__ unsigned short f2bf_bits(float f) {
    unsigned u = __builtin_bit_cast(unsigned, f);
    unsigned r = u + 0x7FFFu + ((u >> 16) & 1u);
    return (unsigned short)(r >> 16);
}

__device__ __forceinline__ float wave_reduce_add(float v) {
    #pragma unroll
    for (int off = 16; off > 0; off >>= 1) v += __shfl_down(v, off, 32);
    return v;
}

// ---------------------------------------------------------------------------
// init: seed encoder & decoder hidden ping-pong buffers with hidden0
// ---------------------------------------------------------------------------
__global__ void k_init(const float* __restrict__ hidden0,
                       float* __restrict__ henc, float* __restrict__ hdec) {
    int i = threadIdx.x;                 // 1024 threads
    float v = hidden0[i];
    henc[i] = v;
    hdec[i] = v;
}

// ---------------------------------------------------------------------------
// convert out_W (fp32, 206MB > L2) -> bf16 (103MB, L2-resident).
// fp32 side is read NON-TEMPORAL (streamed once, must not evict the bf16 copy
// we are installing in L2); bf16 stores use default RT policy so they stick.
// ---------------------------------------------------------------------------
__global__ void k_convert_bf16(const float* __restrict__ outW,
                               unsigned short* __restrict__ wb) {
    size_t g4 = (size_t)blockIdx.x * 256u + threadIdx.x;     // float4 index
    const size_t valid4 = (size_t)VOUT * H / 4;              // divisible by 4
    f32x4 v;
    if (g4 < valid4) {
        v = __builtin_nontemporal_load((const f32x4*)outW + g4);
    } else {
        v = (f32x4)0.0f;                                     // zero pad rows
    }
    u16x4 o;
    o.x = f2bf_bits(v.x); o.y = f2bf_bits(v.y);
    o.z = f2bf_bits(v.z); o.w = f2bf_bits(v.w);
    ((u16x4*)wb)[g4] = o;
}

// ---------------------------------------------------------------------------
// encoder step: h_out = tanh(Wih@x + bih + Whh@h_in + bhh); x = enc_emb[tok]
// wave-per-row GEMV (wave32), 128 blocks x 256 threads = 1024 rows
// ---------------------------------------------------------------------------
__global__ void k_enc_step(const int* __restrict__ inputs, int step,
                           const float* __restrict__ enc_emb,
                           const float* __restrict__ Wih,
                           const float* __restrict__ Whh,
                           const float* __restrict__ bih,
                           const float* __restrict__ bhh,
                           const float* __restrict__ h_in,
                           float* __restrict__ h_out,
                           float* __restrict__ enc_out_row) {
    int lane = threadIdx.x & 31, wid = threadIdx.x >> 5;
    int row = blockIdx.x * 8 + wid;
    int tok = inputs[step];
    const float* x  = enc_emb + (size_t)tok * EDIM;
    const float* wi = Wih + (size_t)row * EDIM;
    const float* wh = Whh + (size_t)row * H;
    float acc = 0.f;
    for (int k = lane; k < EDIM; k += 32) acc += wi[k] * x[k];
    for (int k = lane; k < H;    k += 32) acc += wh[k] * h_in[k];
    acc = wave_reduce_add(acc);
    if (lane == 0) {
        float v = tanhf(acc + bih[row] + bhh[row]);
        h_out[row]       = v;
        enc_out_row[row] = v;
    }
}

// ---------------------------------------------------------------------------
// decoder attention + context (single workgroup, 256 threads)
// aw = softmax(attn_W @ [emb; h] + attn_b);  ctx = aw @ enc_outs
// ---------------------------------------------------------------------------
__global__ void k_attn(const float* __restrict__ dec_emb,
                       const float* __restrict__ attn_W,
                       const float* __restrict__ attn_b,
                       const float* __restrict__ enc_outs,
                       const float* __restrict__ h_cur,
                       float* __restrict__ ctx,
                       const int* __restrict__ token, int step) {
    __shared__ float aw[SEQL];
    __shared__ float m_s, inv_s;
    int lane = threadIdx.x & 31, wid = threadIdx.x >> 5;
    int tok = (step == 0) ? 0 : *token;
    const float* emb = dec_emb + (size_t)tok * H;

    for (int row = wid; row < SEQL; row += 8) {
        const float* ar = attn_W + (size_t)row * (2 * H);
        float acc = 0.f;
        for (int k = lane; k < 2 * H; k += 32) {
            float c = (k < H) ? emb[k] : h_cur[k - H];
            acc += ar[k] * c;
        }
        acc = wave_reduce_add(acc);
        if (lane == 0) aw[row] = acc + attn_b[row];
    }
    __syncthreads();
    if (threadIdx.x == 0) {
        float m = -1e30f;
        for (int l = 0; l < SEQL; ++l) m = fmaxf(m, aw[l]);
        m_s = m;
    }
    __syncthreads();
    if (threadIdx.x < SEQL) aw[threadIdx.x] = __expf(aw[threadIdx.x] - m_s);
    __syncthreads();
    if (threadIdx.x == 0) {
        float s = 0.f;
        for (int l = 0; l < SEQL; ++l) s += aw[l];
        inv_s = 1.f / s;
    }
    __syncthreads();
    for (int j = threadIdx.x; j < H; j += 256) {
        float acc = 0.f;
        #pragma unroll 8
        for (int l = 0; l < SEQL; ++l) acc += aw[l] * enc_outs[(size_t)l * H + j];
        ctx[j] = acc * inv_s;
    }
}

// ---------------------------------------------------------------------------
// o = relu(comb_W @ [emb; ctx] + comb_b)   (wave-per-row, 128 blocks)
// ---------------------------------------------------------------------------
__global__ void k_comb(const float* __restrict__ dec_emb,
                       const float* __restrict__ comb_W,
                       const float* __restrict__ comb_b,
                       const float* __restrict__ ctx,
                       float* __restrict__ o,
                       const int* __restrict__ token, int step) {
    int lane = threadIdx.x & 31, wid = threadIdx.x >> 5;
    int row = blockIdx.x * 8 + wid;
    int tok = (step == 0) ? 0 : *token;
    const float* emb = dec_emb + (size_t)tok * H;
    const float* wr  = comb_W + (size_t)row * (2 * H);
    float acc = 0.f;
    for (int k = lane; k < 2 * H; k += 32) {
        float c = (k < H) ? emb[k] : ctx[k - H];
        acc += wr[k] * c;
    }
    acc = wave_reduce_add(acc);
    if (lane == 0) o[row] = fmaxf(acc + comb_b[row], 0.0f);
}

// ---------------------------------------------------------------------------
// h_new = tanh(dec_Wih@o + dec_bih + dec_Whh@h_old + dec_bhh)
// ---------------------------------------------------------------------------
__global__ void k_hidden(const float* __restrict__ Wih,
                         const float* __restrict__ Whh,
                         const float* __restrict__ bih,
                         const float* __restrict__ bhh,
                         const float* __restrict__ o,
                         const float* __restrict__ h_old,
                         float* __restrict__ h_new) {
    int lane = threadIdx.x & 31, wid = threadIdx.x >> 5;
    int row = blockIdx.x * 8 + wid;
    const float* wi = Wih + (size_t)row * H;
    const float* wh = Whh + (size_t)row * H;
    float acc = 0.f;
    for (int k = lane; k < H; k += 32) acc += wi[k] * o[k] + wh[k] * h_old[k];
    acc = wave_reduce_add(acc);
    if (lane == 0) h_new[row] = tanhf(acc + bih[row] + bhh[row]);
}

// ---------------------------------------------------------------------------
// logits GEMV via WMMA bf16: each wave -> 16 rows of out_W, K=1024 in 32 tiles.
// A = 16x32 bf16 tile of out_W (ISA layout: lanes<16 hold K0-7/K16-23,
// lanes>=16 hold K8-15/K24-31, row = rowbase + (lane&15)).
// B = 32x16 with every column = h (each lane holds h in the same k order as A).
// B is reloaded from LDS hs every tile (two 16-byte loads at the same
// k offsets as A); the preloaded hpre values are not used.
// D column N=0: lane0 c[r]=y[rowbase+r], lane16 c[r]=y[rowbase+8+r].
// 393 blocks x 8 waves x 16 rows = 50304 rows (pad rows are zero weights).
// ---------------------------------------------------------------------------
__global__ void __launch_bounds__(256)
k_logits_wmma(const unsigned short* __restrict__ wb,
              const float* __restrict__ h,
              const float* __restrict__ out_b,
              float* __restrict__ out_row) {
    __shared__ __attribute__((aligned(16))) unsigned short hs[H];
    for (int i = threadIdx.x; i < H; i += 256) hs[i] = f2bf_bits(h[i]);
    __syncthreads();

    int lane = threadIdx.x & 31, wid = threadIdx.x >> 5;
    int rowbase = (blockIdx.x * 8 + wid) * 16;
    const unsigned short* wrow = wb + (size_t)(rowbase + (lane & 15)) * H;
    const int koff = (lane >> 4) * 8;   // half-row K phase for this lane group

    // preload this lane's 32 h values (zero-extended u16 -> {lo=h, hi=0})
    unsigned hpre[32];
    #pragma unroll
    for (int t = 0; t < 32; ++t) hpre[t] = (unsigned)hs[t * 32 + lane];

    v8f c;
    #pragma unroll
    for (int i = 0; i < 8; ++i) c[i] = 0.0f;

    union { uint4 u[2]; v16bf v; } B;

    #pragma unroll
    for (int t = 0; t < 32; ++t) {
        const int kk = t * 32;
        union { uint4 u[2]; v16bf v; } A;
        A.u[0] = *(const uint4*)(wrow + kk + koff);
        A.u[1] = *(const uint4*)(wrow + kk + 16 + koff);
        if (kk + 64 < H) __builtin_prefetch(wrow + kk + 64 + koff, 0, 1);

        B.u[0] = *(const uint4*)(hs + kk + koff);
        B.u[1] = *(const uint4*)(hs + kk + 16 + koff);

        c = __builtin_amdgcn_wmma_f32_16x16x32_bf16(
                false, A.v, false, B.v, (short)0, c, false, false);
    }

    if ((lane & 15) == 0) {
        int rb = rowbase + (lane >> 4) * 8;
        #pragma unroll
        for (int r = 0; r < 8; ++r) {
            int row = rb + r;
            if (row < VOUT) out_row[row] = c[r] + out_b[row];
        }
    }
}

// ---------------------------------------------------------------------------
// fused: argmax (first-index ties), logsumexp, in-place log_softmax,
// write greedy next token. Single workgroup, 1024 threads, wave32 shfl
// reductions first, then one cross-wave pass (32 partials).
// ---------------------------------------------------------------------------
__global__ void __launch_bounds__(1024)
k_logsoftmax_argmax(float* __restrict__ out_row, int* __restrict__ token) {
    __shared__ float wmax[32];
    __shared__ int   widx[32];
    __shared__ float wsum[32];
    __shared__ float g_max_s, g_logZ;
    __shared__ int   g_idx_s;
    int tid = threadIdx.x, lane = tid & 31, wid = tid >> 5;

    // pass 1: max + argmax (first-index tie-break)
    float bm = -1e30f; int bi = 0x7fffffff;
    for (int i = tid; i < VOUT; i += 1024) {
        float v = out_row[i];
        if (v > bm) { bm = v; bi = i; }
    }
    #pragma unroll
    for (int off = 16; off > 0; off >>= 1) {
        float ov = __shfl_down(bm, off, 32);
        int   oi = __shfl_down(bi, off, 32);
        if (ov > bm || (ov == bm && oi < bi)) { bm = ov; bi = oi; }
    }
    if (lane == 0) { wmax[wid] = bm; widx[wid] = bi; }
    __syncthreads();
    if (wid == 0) {
        float v = wmax[lane]; int ix = widx[lane];
        #pragma unroll
        for (int off = 16; off > 0; off >>= 1) {
            float ov = __shfl_down(v, off, 32);
            int   oi = __shfl_down(ix, off, 32);
            if (ov > v || (ov == v && oi < ix)) { v = ov; ix = oi; }
        }
        if (lane == 0) { g_max_s = v; g_idx_s = ix; }
    }
    __syncthreads();
    float gmax = g_max_s;

    // pass 2: logsumexp
    float ls = 0.f;
    for (int i = tid; i < VOUT; i += 1024) ls += __expf(out_row[i] - gmax);
    ls = wave_reduce_add(ls);
    if (lane == 0) wsum[wid] = ls;
    __syncthreads();
    if (wid == 0) {
        float s = wave_reduce_add(wsum[lane]);
        if (lane == 0) g_logZ = gmax + __logf(s);
    }
    __syncthreads();
    float logZ = g_logZ;

    // pass 3: normalize in place; emit greedy token
    for (int i = tid; i < VOUT; i += 1024) out_row[i] -= logZ;
    if (tid == 0) *token = g_idx_s;
}

// ---------------------------------------------------------------------------
// launcher
// ---------------------------------------------------------------------------
extern "C" void kernel_launch(void* const* d_in, const int* in_sizes, int n_in,
                              void* d_out, int out_size, void* d_ws, size_t ws_size,
                              hipStream_t stream) {
    const int*   inputs  = (const int*)  d_in[0];
    const float* hidden0 = (const float*)d_in[2];
    const float* enc_emb = (const float*)d_in[3];
    const float* enc_Wih = (const float*)d_in[4];
    const float* enc_Whh = (const float*)d_in[5];
    const float* enc_bih = (const float*)d_in[6];
    const float* enc_bhh = (const float*)d_in[7];
    const float* dec_emb = (const float*)d_in[8];
    const float* attn_W  = (const float*)d_in[9];
    const float* attn_b  = (const float*)d_in[10];
    const float* comb_W  = (const float*)d_in[11];
    const float* comb_b  = (const float*)d_in[12];
    const float* dec_Wih = (const float*)d_in[13];
    const float* dec_Whh = (const float*)d_in[14];
    const float* dec_bih = (const float*)d_in[15];
    const float* dec_bhh = (const float*)d_in[16];
    const float* out_W   = (const float*)d_in[17];
    const float* out_b   = (const float*)d_in[18];
    float* out = (float*)d_out;

    // workspace layout
    char* ws = (char*)d_ws;
    size_t off = 0;
    unsigned short* wb = (unsigned short*)(ws + off); off += (size_t)VPAD * H * 2;  // 103 MB bf16 out_W
    float* enc_outs = (float*)(ws + off); off += (size_t)SEQL * H * 4;
    float* henc     = (float*)(ws + off); off += 2 * H * 4;   // ping-pong
    float* hdec     = (float*)(ws + off); off += 2 * H * 4;   // ping-pong
    float* ctx      = (float*)(ws + off); off += H * 4;
    float* obuf     = (float*)(ws + off); off += H * 4;
    int*   token    = (int*)  (ws + off); off += 256;
    (void)ws_size; (void)in_sizes; (void)n_in; (void)out_size;

    k_init<<<1, 1024, 0, stream>>>(hidden0, henc, hdec);

    // one streamed (non-temporal) HBM pass over fp32 out_W;
    // bf16 copy is L2-resident for all 64 decoder steps
    k_convert_bf16<<<(VPAD * H / 4) / 256, 256, 0, stream>>>(out_W, wb);

    // sequential encoder (ping-pong hidden)
    for (int s = 0; s < SEQL; ++s) {
        k_enc_step<<<128, 256, 0, stream>>>(inputs, s, enc_emb, enc_Wih, enc_Whh,
                                            enc_bih, enc_bhh,
                                            henc + (s & 1) * H,
                                            henc + ((s + 1) & 1) * H,
                                            enc_outs + (size_t)s * H);
    }

    // sequential greedy decoder (token carried through workspace)
    for (int s = 0; s < SEQL; ++s) {
        float* hold = hdec + (s & 1) * H;
        float* hnew = hdec + ((s + 1) & 1) * H;
        float* orow = out + (size_t)s * VOUT;

        k_attn  <<<1,   256, 0, stream>>>(dec_emb, attn_W, attn_b, enc_outs,
                                          hold, ctx, token, s);
        k_comb  <<<128, 256, 0, stream>>>(dec_emb, comb_W, comb_b, ctx, obuf,
                                          token, s);
        k_hidden<<<128, 256, 0, stream>>>(dec_Wih, dec_Whh, dec_bih, dec_bhh,
                                          obuf, hold, hnew);
        k_logits_wmma<<<VPAD / 128, 256, 0, stream>>>(wb, hnew, out_b, orow);
        k_logsoftmax_argmax<<<1, 1024, 0, stream>>>(orow, token);
    }
}
